// SpatialAttention_60748017434794
// MI455X (gfx1250) — matrix operand order repaired and checked by reading, not compiled, not run on hardware
//
#include <hip/hip_runtime.h>

typedef __attribute__((ext_vector_type(16))) __bf16 v16bf;
typedef __attribute__((ext_vector_type(8)))  float  v8f;

#define NHc 8
#define Vc  25

// workspace layout (bytes), all 16B aligned
#define WS_KEFF   0        // ushort[625*32]  = 40000 B
#define WS_BIAS   40960    // float[625]      = 2500 B
#define WS_QKVW   43520    // ushort[768*256] = 393216 B
#define WS_ATTNW  436736   // ushort[256*256] = 131072 B

// ---------- bf16 bit helpers (no __bf16 scalar math) ----------
__device__ inline unsigned short f2bf(float f) {
  unsigned u = __float_as_uint(f);
  u = u + 0x7fffu + ((u >> 16) & 1u);   // round-to-nearest-even
  return (unsigned short)(u >> 16);
}
__device__ inline float2 bfpair(unsigned u) {
  float2 r;
  r.x = __uint_as_float(u << 16);
  r.y = __uint_as_float(u & 0xffff0000u);
  return r;
}

union FragBF { uint4 u[2]; v16bf v; };

// A-matrix 16x32 bf16 fragment: 16 contiguous K values at p
// (caller pre-applies the (lane<16 ? 0 : 16) K offset)
__device__ inline v16bf fragA(const unsigned short* p) {
  FragBF f;
  f.u[0] = *(const uint4*)(p);
  f.u[1] = *(const uint4*)(p + 8);
  return f.v;
}
// B-matrix 32x16 bf16 fragment: 16 contiguous K values at p
// (caller pre-applies the (lane<16 ? 0 : 16) K offset)
__device__ inline v16bf fragB(const unsigned short* p) {
  FragBF f;
  f.u[0] = *(const uint4*)(p);
  f.u[1] = *(const uint4*)(p + 8);
  return f.v;
}

__device__ inline float dot32bf(const unsigned short* a, const unsigned short* b) {
  const uint4* ua = (const uint4*)a;
  const uint4* ub = (const uint4*)b;
  float s = 0.0f;
#pragma unroll
  for (int i = 0; i < 4; ++i) {
    uint4 va = ua[i], vb = ub[i];
    unsigned aw[4] = {va.x, va.y, va.z, va.w};
    unsigned bw[4] = {vb.x, vb.y, vb.z, vb.w};
#pragma unroll
    for (int j = 0; j < 4; ++j) {
      float2 fa = bfpair(aw[j]);
      float2 fb = bfpair(bw[j]);
      s = fmaf(fa.x, fb.x, s);
      s = fmaf(fa.y, fb.y, s);
    }
  }
  return s;
}

#define WMMA_BF16(A, B, C) \
  __builtin_amdgcn_wmma_f32_16x16x32_bf16(false, (A), false, (B), (short)0, (C), false, false)

// ---------------- prep: Keff table, bias, bf16 weight packs ----------------
__global__ void spatial_attn_prep(const float* __restrict__ qkv_w,
                                  const float* __restrict__ attn_w,
                                  const float* __restrict__ A,
                                  const float* __restrict__ mask,
                                  const float* __restrict__ key_rel,
                                  const float* __restrict__ key_rel_diag,
                                  unsigned char* __restrict__ ws) {
  unsigned short* keff = (unsigned short*)(ws + WS_KEFF);
  float*          bias = (float*)(ws + WS_BIAS);
  unsigned short* qw   = (unsigned short*)(ws + WS_QKVW);
  unsigned short* aw   = (unsigned short*)(ws + WS_ATTNW);
  int t = threadIdx.x;
  // rel_to_abs closed form: n==m -> diag; n>m -> key_rel[m*24+n-1]; n<m -> key_rel[m*24+n]
  for (int p = t; p < Vc * Vc; p += 256) {
    int m = p / Vc, n = p % Vc;
    bias[p] = A[p] * mask[p];
    const float* src;
    if (n == m)      src = key_rel_diag;
    else if (n > m)  src = key_rel + (m * 24 + (n - 1)) * 32;
    else             src = key_rel + (m * 24 + n) * 32;
#pragma unroll
    for (int d = 0; d < 32; ++d) keff[p * 32 + d] = f2bf(src[d]);
  }
  for (int i = t; i < 768 * 256; i += 256) qw[i] = f2bf(qkv_w[i]);
  for (int i = t; i < 256 * 256; i += 256) aw[i] = f2bf(attn_w[i]);
}

// ---------------- fused main: one workgroup per batch element ----------------
__global__ __launch_bounds__(256)
void spatial_attn_main(const float* __restrict__ x,
                       const float* __restrict__ qkv_b,
                       const float* __restrict__ attn_b,
                       const unsigned char* __restrict__ ws,
                       float* __restrict__ out) {
  const unsigned short* keff = (const unsigned short*)(ws + WS_KEFF);
  const float*          bias = (const float*)(ws + WS_BIAS);
  const unsigned short* qw   = (const unsigned short*)(ws + WS_QKVW);
  const unsigned short* aw   = (const unsigned short*)(ws + WS_ATTNW);

  // 64 KB LDS total
  __shared__ __align__(16) unsigned short sX[32 * 256];      // phase A: x[node][c]; phase C/D: attn-out[node][c]
  __shared__ __align__(16) unsigned short sQ[NHc * 32 * 32]; // q scaled, [h][node][d]
  __shared__ __align__(16) unsigned short sK[NHc * 32 * 32]; // k [h][node][d]; later softmax-w [h][m][n]
  __shared__ __align__(16) unsigned short sV[NHc * 32 * 32]; // v [h][d][node]

  const int b    = blockIdx.x;
  const int tid  = threadIdx.x;
  const int wave = tid >> 5;
  const int lane = tid & 31;
  const int l16  = lane & 15;
  const int hiH  = (lane >> 4);          // 0 or 1 (half-wave)
  const int kA0  = hiH ? 16 : 0;         // A-fragment K offset
  const int kB0  = hiH ? 16 : 0;         // B-fragment K offset

  // ---- load x (256 ch x 25 nodes, fp32) -> sX[node][c] bf16, zero-pad nodes 25..31
  {
    const float* xb = x + (size_t)b * (256 * Vc);
    for (int e = tid; e < 32 * 256; e += 256) {
      int node = e >> 8, c = e & 255;
      float v = (node < Vc) ? xb[c * Vc + node] : 0.0f;
      sX[e] = f2bf(v);
    }
  }
  __syncthreads();

  // ================= Phase A: qkv = W(768x256) @ x(256x32) =================
  {
    v8f acc[6][2];
#pragma unroll
    for (int i = 0; i < 6; ++i)
#pragma unroll
      for (int j = 0; j < 2; ++j)
#pragma unroll
        for (int r = 0; r < 8; ++r) acc[i][j][r] = 0.0f;

#pragma unroll
    for (int kb = 0; kb < 8; ++kb) {
      const int k0 = kb * 32;
      v16bf b0 = fragB(sX + (l16 + 0)  * 256 + k0 + kB0);
      v16bf b1 = fragB(sX + (l16 + 16) * 256 + k0 + kB0);
#pragma unroll
      for (int i = 0; i < 6; ++i) {
        const int mt = wave * 6 + i;
        v16bf a = fragA(qw + (mt * 16 + l16) * 256 + k0 + kA0);
        acc[i][0] = WMMA_BF16(a, b0, acc[i][0]);
        acc[i][1] = WMMA_BF16(a, b1, acc[i][1]);
      }
    }
    const float qscale = 0.17677669529663687f; // dkh^-0.5
#pragma unroll
    for (int i = 0; i < 6; ++i) {
      const int mt = wave * 6 + i;
#pragma unroll
      for (int nt = 0; nt < 2; ++nt) {
        const int n = l16 + nt * 16;
#pragma unroll
        for (int r = 0; r < 8; ++r) {
          const int o = mt * 16 + r + (hiH ? 8 : 0);
          const float v = acc[i][nt][r] + qkv_b[o];
          if (o < 256) {                      // q (scaled)
            int h = o >> 5, d = o & 31;
            sQ[h * 1024 + n * 32 + d] = f2bf(v * qscale);
          } else if (o < 512) {               // k
            int o2 = o - 256; int h = o2 >> 5, d = o2 & 31;
            sK[h * 1024 + n * 32 + d] = f2bf(v);
          } else {                            // v
            int o2 = o - 512; int h = o2 >> 5, d = o2 & 31;
            sV[h * 1024 + d * 32 + n] = f2bf(v);
          }
        }
      }
    }
  }
  __syncthreads();

  // ================= Phase B: per-head logits + softmax (wave == head) =====
  const int h = wave;
  const unsigned short* qh = sQ + h * 1024;
  const unsigned short* kh = sK + h * 1024;
  const unsigned short* vh = sV + h * 1024;
  unsigned short*      wst = sK + h * 1024;  // alias: reuse k block for softmax weights

  v8f lg[2][2];
  // init C with adjacency bias + relative-position logits (Keff table, WGP$-resident)
#pragma unroll
  for (int mt = 0; mt < 2; ++mt) {
#pragma unroll
    for (int r = 0; r < 8; ++r) {
      const int m = mt * 16 + r + (hiH ? 8 : 0);
#pragma unroll
      for (int nt = 0; nt < 2; ++nt) {
        const int n = l16 + nt * 16;
        float cv = 0.0f;
        if (m < Vc && n < Vc)
          cv = bias[m * Vc + n] + dot32bf(qh + m * 32, keff + (m * Vc + n) * 32);
        lg[mt][nt][r] = cv;
      }
    }
  }
  // logits += q^T k   (A rows = query node, K = d = 32)
#pragma unroll
  for (int mt = 0; mt < 2; ++mt) {
    v16bf a = fragA(qh + (mt * 16 + l16) * 32 + kA0);
#pragma unroll
    for (int nt = 0; nt < 2; ++nt) {
      v16bf bb = fragB(kh + (l16 + nt * 16) * 32 + kB0);
      lg[mt][nt] = WMMA_BF16(a, bb, lg[mt][nt]);
    }
  }
  // mask padded key columns (n >= 25), then row softmax over n
  if (l16 >= 9) {
#pragma unroll
    for (int mt = 0; mt < 2; ++mt)
#pragma unroll
      for (int r = 0; r < 8; ++r) lg[mt][1][r] = -1e30f;
  }
#pragma unroll
  for (int mt = 0; mt < 2; ++mt) {
#pragma unroll
    for (int r = 0; r < 8; ++r) {
      float mx = fmaxf(lg[mt][0][r], lg[mt][1][r]);
#pragma unroll
      for (int s = 1; s < 16; s <<= 1) mx = fmaxf(mx, __shfl_xor(mx, s, 32));
      float e0 = __expf(lg[mt][0][r] - mx);
      float e1 = __expf(lg[mt][1][r] - mx);
      float sm = e0 + e1;
#pragma unroll
      for (int s = 1; s < 16; s <<= 1) sm += __shfl_xor(sm, s, 32);
      float inv = __frcp_rn(sm);
      lg[mt][0][r] = e0 * inv;
      lg[mt][1][r] = e1 * inv;
    }
  }
  // stage softmax weights into A-matrix layout [m][n] (padded n carry exact 0)
#pragma unroll
  for (int mt = 0; mt < 2; ++mt)
#pragma unroll
    for (int nt = 0; nt < 2; ++nt)
#pragma unroll
      for (int r = 0; r < 8; ++r) {
        const int m = mt * 16 + r + (hiH ? 8 : 0);
        const int n = l16 + nt * 16;
        wst[m * 32 + n] = f2bf(lg[mt][nt][r]);
      }

  // ================= Phase C: out = w(25x32n) @ v^T(32n x 32d) =============
  {
    v8f oc[2][2];
#pragma unroll
    for (int i = 0; i < 2; ++i)
#pragma unroll
      for (int j = 0; j < 2; ++j)
#pragma unroll
        for (int r = 0; r < 8; ++r) oc[i][j][r] = 0.0f;
#pragma unroll
    for (int mt = 0; mt < 2; ++mt) {
      v16bf a = fragA(wst + (mt * 16 + l16) * 32 + kA0);
#pragma unroll
      for (int nt = 0; nt < 2; ++nt) {
        v16bf bb = fragB(vh + (l16 + nt * 16) * 32 + kB0); // col = dv, K = n contiguous
        oc[mt][nt] = WMMA_BF16(a, bb, oc[mt][nt]);
      }
    }
    // write attn output to sX as [node][channel], channel = h*32 + dv
#pragma unroll
    for (int mt = 0; mt < 2; ++mt)
#pragma unroll
      for (int nt = 0; nt < 2; ++nt)
#pragma unroll
        for (int r = 0; r < 8; ++r) {
          const int m  = mt * 16 + r + (hiH ? 8 : 0);
          const int dv = l16 + nt * 16;
          sX[m * 256 + h * 32 + dv] = f2bf(oc[mt][nt][r]);
        }
  }
  __syncthreads();

  // ================= Phase D: out_proj (256x256) @ attn(256x32) + bias =====
  {
    v8f acc[2][2];
#pragma unroll
    for (int i = 0; i < 2; ++i)
#pragma unroll
      for (int j = 0; j < 2; ++j)
#pragma unroll
        for (int r = 0; r < 8; ++r) acc[i][j][r] = 0.0f;

#pragma unroll
    for (int kb = 0; kb < 8; ++kb) {
      const int k0 = kb * 32;
      v16bf b0 = fragB(sX + (l16 + 0)  * 256 + k0 + kB0);
      v16bf b1 = fragB(sX + (l16 + 16) * 256 + k0 + kB0);
#pragma unroll
      for (int i = 0; i < 2; ++i) {
        const int mt = wave * 2 + i;
        v16bf a = fragA(aw + (mt * 16 + l16) * 256 + k0 + kA0);
        acc[i][0] = WMMA_BF16(a, b0, acc[i][0]);
        acc[i][1] = WMMA_BF16(a, b1, acc[i][1]);
      }
    }
    float* ob = out + (size_t)b * (256 * Vc);
#pragma unroll
    for (int i = 0; i < 2; ++i) {
      const int mt = wave * 2 + i;
#pragma unroll
      for (int nt = 0; nt < 2; ++nt) {
        const int n = l16 + nt * 16;
        if (n < Vc) {
#pragma unroll
          for (int r = 0; r < 8; ++r) {
            const int o = mt * 16 + r + (hiH ? 8 : 0);
            ob[o * Vc + n] = acc[i][nt][r] + attn_b[o];
          }
        }
      }
    }
  }
}

extern "C" void kernel_launch(void* const* d_in, const int* in_sizes, int n_in,
                              void* d_out, int out_size, void* d_ws, size_t ws_size,
                              hipStream_t stream) {
  const float* x            = (const float*)d_in[0];
  const float* qkv_w        = (const float*)d_in[1];
  const float* qkv_b        = (const float*)d_in[2];
  const float* attn_w       = (const float*)d_in[3];
  const float* attn_b       = (const float*)d_in[4];
  const float* A            = (const float*)d_in[5];
  const float* mask         = (const float*)d_in[6];
  const float* key_rel      = (const float*)d_in[7];
  const float* key_rel_diag = (const float*)d_in[8];
  float* outp        = (float*)d_out;
  unsigned char* ws  = (unsigned char*)d_ws;

  spatial_attn_prep<<<1, 256, 0, stream>>>(qkv_w, attn_w, A, mask, key_rel, key_rel_diag, ws);
  spatial_attn_main<<<4096, 256, 0, stream>>>(x, qkv_b, attn_b, ws, outp);
}
